// NNFMLoss_5291399709167
// MI455X (gfx1250) — matrix operand order repaired and checked by reading, not compiled, not run on hardware
//
#include <hip/hip_runtime.h>
#include <cstdint>
#include <cstddef>

// ---------------------------------------------------------------------------
// NNFM loss for MI455X (gfx1250).  Dominant cost: 16384x16384x256 GEMM with
// fused argmax -> v_wmma_f32_16x16x32_bf16.  Staging via async global->LDS
// copies (ASYNCcnt).  Final loss recomputed in fp32 at the argmax indices.
//
// v2: each wave owns TWO 16-row M-tiles sharing every B fragment (2 WMMAs per
//     B load -> halved LDS traffic per WMMA) and B fragments are explicitly
//     double-buffered across K-steps so ds_loads overlap the WMMA chain.
// ---------------------------------------------------------------------------

#define EPSF 1e-8f
constexpr int C      = 256;      // channels
constexpr int PQ     = 16384;    // pixels (128*128)
constexpr int BLK    = 128;      // p-block per WG, q-block per outer iter
constexpr int PITCH  = 264;      // bf16 elements per padded LDS row
constexpr int ROWB   = PITCH * 2;      // 528 bytes (33*16 -> b128 aligned)
constexpr int KSTEPS = C / 32;   // 8 WMMA K-steps of 32
constexpr int NTHREADS = 128;    // 4 waves; wave owns 32 p's

typedef __bf16 bf16;
typedef __attribute__((ext_vector_type(16))) __bf16 v16bf;
typedef __attribute__((ext_vector_type(8)))  float  v8f;

union Frag {
    int4  q[2];     // two b128 LDS loads
    v16bf v;        // WMMA operand view
};

// ---- CDNA5 async global->LDS copy (per-lane b128), tracked by ASYNCcnt ----
__device__ __forceinline__ void async_ld_b128(uint32_t lds_off,
                                              const void* sgpr_base,
                                              uint32_t voff)
{
    // vdst = LDS byte address, vaddr = 32-bit offset, saddr = 64-bit base
    asm volatile("global_load_async_to_lds_b128 %0, %1, %2"
                 :: "v"(lds_off), "v"(voff), "s"(sgpr_base)
                 : "memory");
}
__device__ __forceinline__ void wait_async0()
{
    asm volatile("s_wait_asynccnt 0" ::: "memory");
}

// B fragment for K-step ks of a given q row (32x16 bf16, ISA 7.12.2):
// lane = N (q); lanes>=16 take K+8.  V0-3: K=klo..klo+7 (b128),
// V4-7: K=klo+16..klo+23 (b128) -> 2 x ds_load_b128.
__device__ __forceinline__ Frag load_bfrag(const char* brow, int ks, int half)
{
    const int klo = ks * 32 + half * 8;
    Frag f;
    f.q[0] = *(const int4*)(brow + klo * 2);
    f.q[1] = *(const int4*)(brow + klo * 2 + 32);
    return f;
}

// ---------------------------------------------------------------------------
// Kernel 1: build bf16 transposed copies At[p][c], Bt[q][c] and invbn[q].
// ---------------------------------------------------------------------------
__global__ void __launch_bounds__(256) k_prep(const float* __restrict__ a,
                                              const float* __restrict__ b,
                                              bf16* __restrict__ At,
                                              bf16* __restrict__ Bt,
                                              float* __restrict__ invbn)
{
    const int gid  = blockIdx.x * blockDim.x + threadIdx.x;  // 0 .. 2*PQ-1
    const bool isA = gid >= PQ;
    const int col  = gid & (PQ - 1);
    const float* __restrict__ src = isA ? a : b;
    bf16* dst = (isA ? At : Bt) + (size_t)col * C;

    float ss = 0.f;
    for (int c0 = 0; c0 < C; c0 += 8) {
        union { bf16 h[8]; int4 i; } tmp;
#pragma unroll
        for (int i = 0; i < 8; ++i) {
            float v = src[(size_t)(c0 + i) * PQ + col];
            ss += v * v;
            tmp.h[i] = (bf16)v;   // RNE convert
        }
        *reinterpret_cast<int4*>(dst + c0) = tmp.i;
    }
    if (!isA)
        invbn[col] = 1.0f / (sqrtf(ss + EPSF) + EPSF);
}

// ---------------------------------------------------------------------------
// Kernel 2: fused GEMM + argmax.
//   WG = 128 threads = 4 waves; wave w owns p rows [w*32, w*32+32) as two
//   16-row M-tiles.  Outer loop over 128-wide q blocks (async-staged).
//   Inner: per 16-q subtile, 8 K-steps x 2 WMMAs sharing each B fragment.
// ---------------------------------------------------------------------------
__global__ void __launch_bounds__(NTHREADS) k_argmax(const bf16* __restrict__ At,
                                                     const bf16* __restrict__ Bt,
                                                     const float* __restrict__ invbn,
                                                     int* __restrict__ bestq)
{
    extern __shared__ __align__(16) char smem[];
    bf16*  ldsA   = (bf16*)smem;                                  // BLK x PITCH
    bf16*  ldsB   = (bf16*)(smem + (size_t)BLK * ROWB);           // BLK x PITCH
    float* ldsInv = (float*)(smem + (size_t)2 * BLK * ROWB);      // BLK floats

    const uint32_t ldsA_off   = (uint32_t)(uintptr_t)ldsA;
    const uint32_t ldsB_off   = (uint32_t)(uintptr_t)ldsB;
    const uint32_t ldsInv_off = (uint32_t)(uintptr_t)ldsInv;

    const int tid    = threadIdx.x;
    const int wave   = tid >> 5;           // 0..3
    const int lane   = tid & 31;
    const int l16    = lane & 15;
    const int half   = lane >> 4;
    const int pblock = blockIdx.x * BLK;

    // ---- stage A block: 128 rows x 512 B (contiguous 64 KB in global) ----
    {
        const bf16* Abase = At + (size_t)pblock * C;   // uniform -> SGPR pair
#pragma unroll
        for (int k = 0; k < 32; ++k) {
            int idx = tid + NTHREADS * k;          // 4096 b128 chunks
            int row = idx >> 5, chunk = idx & 31;  // 32 chunks per 512 B row
            async_ld_b128(ldsA_off + row * ROWB + chunk * 16, Abase,
                          (uint32_t)idx * 16);
        }
    }

    float bestS0[8], bestS1[8];
    int   bestQ0[8], bestQ1[8];
#pragma unroll
    for (int r = 0; r < 8; ++r) {
        bestS0[r] = -3.4e38f; bestQ0[r] = 0;
        bestS1[r] = -3.4e38f; bestQ1[r] = 0;
    }

    v16bf afrag0[KSTEPS];   // M-tile 0: rows wave*32 + l16       (64 VGPRs)
    v16bf afrag1[KSTEPS];   // M-tile 1: rows wave*32 + 16 + l16  (64 VGPRs)

    for (int j = 0; j < PQ / BLK; ++j) {
        const int qblock = j * BLK;
        __syncthreads();   // previous B tile fully consumed

        const bf16* Bbase = Bt + (size_t)qblock * C;
#pragma unroll
        for (int k = 0; k < 32; ++k) {
            int idx = tid + NTHREADS * k;
            int row = idx >> 5, chunk = idx & 31;
            async_ld_b128(ldsB_off + row * ROWB + chunk * 16, Bbase,
                          (uint32_t)idx * 16);
        }
        if (tid < 32)   // 32 lanes * 16 B = 128 floats of invbn
            async_ld_b128(ldsInv_off + tid * 16, invbn + qblock,
                          (uint32_t)tid * 16);
        wait_async0();
        __syncthreads();

        if (j == 0) {
            // A fragment layout (16-bit A 16x32, ISA 7.12.2):
            //   lane = M (p); lanes>=16 take K+8.  V0-3: K=klo..klo+7 (b128),
            //   V4-7: K=klo+16..klo+23 (b128).
            const char* arow0 = (const char*)ldsA + (wave * 32 + l16) * ROWB;
            const char* arow1 = arow0 + 16 * ROWB;
#pragma unroll
            for (int ks = 0; ks < KSTEPS; ++ks) {
                const int klo = ks * 32 + half * 8;
                Frag f;
                f.q[0] = *(const int4*)(arow0 + klo * 2);
                f.q[1] = *(const int4*)(arow0 + klo * 2 + 32);
                afrag0[ks] = f.v;
                f.q[0] = *(const int4*)(arow1 + klo * 2);
                f.q[1] = *(const int4*)(arow1 + klo * 2 + 32);
                afrag1[ks] = f.v;
            }
        }

        for (int t = 0; t < 8; ++t) {
            const int   qrow = t * 16 + l16;
            const char* brow = (const char*)ldsB + qrow * ROWB;

            v8f acc0 = {};
            v8f acc1 = {};
            // double-buffered B fragments: loads for ks+1 issue before the
            // two WMMAs of ks, so ds latency hides under the XDL pipe.
            Frag cur = load_bfrag(brow, 0, half);
#pragma unroll
            for (int ks = 0; ks < KSTEPS; ++ks) {
                Frag nxt;
                if (ks + 1 < KSTEPS) nxt = load_bfrag(brow, ks + 1, half);
                acc0 = __builtin_amdgcn_wmma_f32_16x16x32_bf16(
                         false, afrag0[ks], false, cur.v,
                         (short)0, acc0, false, false);
                acc1 = __builtin_amdgcn_wmma_f32_16x16x32_bf16(
                         false, afrag1[ks], false, cur.v,
                         (short)0, acc1, false, false);
                if (ks + 1 < KSTEPS) cur = nxt;
            }

            const float inv = ldsInv[qrow];
            const int   q   = qblock + qrow;
#pragma unroll
            for (int r = 0; r < 8; ++r) {
                // C/D layout: VGPR r -> M = r + 8*half; lane16 -> N.
                float s0 = acc0[r] * inv;
                if (s0 > bestS0[r]) { bestS0[r] = s0; bestQ0[r] = q; }
                float s1 = acc1[r] * inv;
                if (s1 > bestS1[r]) { bestS1[r] = s1; bestQ1[r] = q; }
            }
        }
    }

    // ---- merge argmax across the 16 lanes sharing each p ----
#pragma unroll
    for (int r = 0; r < 8; ++r) {
        float s = bestS0[r];
        int   q = bestQ0[r];
        for (int off = 8; off > 0; off >>= 1) {
            float s2 = __shfl_xor(s, off, 16);
            int   q2 = __shfl_xor(q, off, 16);
            if (s2 > s || (s2 == s && q2 < q)) { s = s2; q = q2; }
        }
        if (l16 == 0)
            bestq[pblock + wave * 32 + half * 8 + r] = q;

        s = bestS1[r];
        q = bestQ1[r];
        for (int off = 8; off > 0; off >>= 1) {
            float s2 = __shfl_xor(s, off, 16);
            int   q2 = __shfl_xor(q, off, 16);
            if (s2 > s || (s2 == s && q2 < q)) { s = s2; q = q2; }
        }
        if (l16 == 0)
            bestq[pblock + wave * 32 + 16 + half * 8 + r] = q;
    }
}

// ---------------------------------------------------------------------------
// Kernel 3: fp32-exact loss at the chosen indices (b is L2-resident gather).
// ---------------------------------------------------------------------------
__global__ void __launch_bounds__(256) k_exact(const float* __restrict__ a,
                                               const float* __restrict__ b,
                                               const int* __restrict__ bestq,
                                               float* __restrict__ lossv)
{
    const int p = blockIdx.x * blockDim.x + threadIdx.x;
    const int q = bestq[p];
    float dot = 0.f, na = 0.f, nb = 0.f;
    for (int c = 0; c < C; ++c) {
        float av = a[(size_t)c * PQ + p];
        float bv = b[(size_t)c * PQ + q];
        dot += av * bv;
        na  += av * av;
        nb  += bv * bv;
    }
    lossv[p] = 1.f - dot / ((sqrtf(na) + EPSF) * (sqrtf(nb) + EPSF));
}

// ---------------------------------------------------------------------------
// Kernel 4: deterministic fixed-order reduction to the scalar mean.
// ---------------------------------------------------------------------------
__global__ void __launch_bounds__(256) k_reduce(const float* __restrict__ lossv,
                                                float* __restrict__ out)
{
    __shared__ float s[256];
    const int tid = threadIdx.x;
    float sum = 0.f;
    for (int i = 0; i < PQ / 256; ++i)
        sum += lossv[tid + 256 * i];
    s[tid] = sum;
    __syncthreads();
    for (int w = 128; w > 0; w >>= 1) {
        if (tid < w) s[tid] += s[tid + w];
        __syncthreads();
    }
    if (tid == 0) out[0] = s[0] / (float)PQ;
}

// ---------------------------------------------------------------------------
extern "C" void kernel_launch(void* const* d_in, const int* in_sizes, int n_in,
                              void* d_out, int out_size, void* d_ws, size_t ws_size,
                              hipStream_t stream)
{
    const float* a = (const float*)d_in[0];
    const float* b = (const float*)d_in[1];
    float* out = (float*)d_out;

    char* ws = (char*)d_ws;
    bf16*  At    = (bf16*)ws;                                        // 8 MB
    bf16*  Bt    = (bf16*)(ws + (size_t)PQ * C * 2);                 // 8 MB
    float* invbn = (float*)(ws + (size_t)PQ * C * 4);                // 64 KB
    int*   bestq = (int*)  (ws + (size_t)PQ * C * 4 + (size_t)PQ * 4);
    float* lossv = (float*)(ws + (size_t)PQ * C * 4 + (size_t)PQ * 8);

    k_prep  <<<2 * PQ / 256, 256, 0, stream>>>(a, b, At, Bt, invbn);

    const size_t smem = (size_t)2 * BLK * ROWB + BLK * sizeof(float);
    k_argmax<<<PQ / BLK, NTHREADS, smem, stream>>>(At, Bt, invbn, bestq);

    k_exact <<<PQ / 256, 256, 0, stream>>>(a, b, bestq, lossv);
    k_reduce<<<1, 256, 0, stream>>>(lossv, out);
}
